// UMNNModel_45294725104115
// MI455X (gfx1250) — matrix operand order repaired and checked by reading, not compiled, not run on hardware
//
#include <hip/hip_runtime.h>
#include <hip/hip_bf16.h>

typedef __attribute__((ext_vector_type(16))) _Float16 v16h;
typedef __attribute__((ext_vector_type(8)))  _Float16 v8h;
typedef __attribute__((ext_vector_type(8)))  float    v8f;

#define BDIM 4096
#define INF_ 64
#define KF   16
#define SF   32
#define HF   64
#define NMF  48
#define NHF  512

static __device__ __forceinline__ v8f wmma_f16(v16h a, v16h b, v8f c) {
  return __builtin_amdgcn_wmma_f32_16x16x32_f16(false, a, false, b, (short)0, c,
                                                false, false);
}

// ---- native TRANS32 activations (single-instruction on gfx1250) -----------
static __device__ __forceinline__ float fast_tanh(float x) {
#if __has_builtin(__builtin_amdgcn_tanhf)
  return __builtin_amdgcn_tanhf(x);
#else
  float r;
  asm volatile("v_tanh_f32 %0, %1\n\tv_nop" : "=v"(r) : "v"(x));
  return r;
#endif
}
static __device__ __forceinline__ float fast_exp2(float x) {
#if __has_builtin(__builtin_amdgcn_exp2f)
  return __builtin_amdgcn_exp2f(x);
#else
  float r;
  asm volatile("v_exp_f32 %0, %1\n\tv_nop" : "=v"(r) : "v"(x));
  return r;
#endif
}
static __device__ __forceinline__ float fast_log2(float x) {
#if __has_builtin(__builtin_amdgcn_logf)
  return __builtin_amdgcn_logf(x);
#else
  float r;
  asm volatile("v_log_f32 %0, %1\n\tv_nop" : "=v"(r) : "v"(x));
  return r;
#endif
}
// softplus(z) = ln2 * log2(1 + 2^(z*log2e)); branchless large-z guard
static __device__ __forceinline__ float fast_softplus(float z) {
  float e = fast_exp2(z * 1.44269504f);
  float g = 0.69314718f * fast_log2(1.0f + e);
  return (z > 20.f) ? z : g;
}

// ---------------------------------------------------------------------------
// Prep: transpose + convert weights to f16 in B-fragment-friendly (col-major)
// layouts. mw2t[k][j][h], nw1t[j][kk(pad64)], nw2t[j][K].
// ---------------------------------------------------------------------------
__global__ void prep_kernel(const float* __restrict__ mw2,
                            const float* __restrict__ nw1,
                            const float* __restrict__ nw2,
                            _Float16* __restrict__ mw2t,
                            _Float16* __restrict__ nw1t,
                            _Float16* __restrict__ nw2t) {
  int i = blockIdx.x * blockDim.x + threadIdx.x;
  const int N1 = KF * HF * HF;       // 65536
  const int N2 = NHF * 64;           // 32768 (K padded 48->64)
  const int N3 = NHF * NHF;          // 262144
  if (i < N1) {
    int h = i & 63, j = (i >> 6) & 63, k = i >> 12;
    mw2t[i] = (_Float16)mw2[((size_t)(k * HF + h)) * HF + j];
  } else if (i < N1 + N2) {
    int q = i - N1;
    int kk = q & 63, j = q >> 6;
    nw1t[q] = (kk < NMF) ? (_Float16)nw1[(size_t)kk * NHF + j] : (_Float16)0.f;
  } else if (i < N1 + N2 + N3) {
    int q = i - (N1 + N2);
    int K = q & 511, j = q >> 9;
    nw2t[q] = (_Float16)nw2[(size_t)K * NHF + j];
  }
}

// ---------------------------------------------------------------------------
// Mono path: for each k, GEMM (B*S x 64) @ mw2[k](64x64) with native-tanh
// producer (h1) and native tanh/softplus consumer. One wave = one 16-row
// tile (fixed b, 16 consecutive s). Writes per-(b,k,shalf) softplus sums.
// ---------------------------------------------------------------------------
__global__ __launch_bounds__(256) void mono_kernel(
    const float* __restrict__ x, const float* __restrict__ mw1,
    const float* __restrict__ mb1, const float* __restrict__ mb2,
    const float* __restrict__ mw3, const float* __restrict__ mb3,
    const _Float16* __restrict__ mw2t, float* __restrict__ gsum) {
  const int k    = blockIdx.y;
  const int lane = threadIdx.x & 31;
  const int wave = threadIdx.x >> 5;
  const int lh   = lane >> 4;   // half: 0 or 1
  const int l15  = lane & 15;

  // --- hoisted per-k data: mw1/mb1 at this lane's 32 A-fragment K indices
  float w1r[32], b1r[32];
#pragma unroll
  for (int c = 0; c < 2; ++c)
#pragma unroll
    for (int g = 0; g < 2; ++g) {
      int kkb = 32 * c + 16 * lh + 8 * g;
#pragma unroll
      for (int i2 = 0; i2 < 8; ++i2) {
        w1r[c * 16 + g * 8 + i2] = mw1[(size_t)k * HF + kkb + i2];
        b1r[c * 16 + g * 8 + i2] = mb1[(size_t)k * HF + kkb + i2];
      }
    }
  // B fragments of mw2[k]: lane holds column j = 16*nt + l15
  v16h Bf[4][2];
  float mb2r[4], mw3r[4];
#pragma unroll
  for (int nt = 0; nt < 4; ++nt) {
    int j = 16 * nt + l15;
#pragma unroll
    for (int c = 0; c < 2; ++c)
      Bf[nt][c] =
          *(const v16h*)(mw2t + ((size_t)(k * HF + j)) * HF + 32 * c + 16 * lh);
    mb2r[nt] = mb2[(size_t)k * HF + j];
    mw3r[nt] = mw3[(size_t)k * HF + j];
  }
  const float mb3k = mb3[k];

  // --- tile loop ----------------------------------------------------------
  const int TPW = 8;  // 128 blocks * 8 waves * 8 = 8192 tiles per k
  int wave_global = blockIdx.x * 8 + wave;
  for (int it = 0; it < TPW; ++it) {
    int t  = wave_global * TPW + it;   // [0, 8192)
    int b  = t >> 1;
    int s0 = (t & 1) * 16;
    float xm  = x[(size_t)b * INF_ + k];
    float pts = xm * (float)(s0 + l15) * (1.0f / 31.0f);  // row m = l15

    v8f acc[4];
#pragma unroll
    for (int nt = 0; nt < 4; ++nt) {
      v8f z = {0.f, 0.f, 0.f, 0.f, 0.f, 0.f, 0.f, 0.f};
      acc[nt] = z;
    }
#pragma unroll
    for (int c = 0; c < 2; ++c) {
      // h1 A-fragment computed on the fly with native v_tanh_f32
      v16h a;
#pragma unroll
      for (int e = 0; e < 16; ++e) {
        float v = fast_tanh(fmaf(pts, w1r[c * 16 + e], b1r[c * 16 + e]));
        a[e] = (_Float16)v;
      }
#pragma unroll
      for (int nt = 0; nt < 4; ++nt) acc[nt] = wmma_f16(a, Bf[nt][c], acc[nt]);
    }
    // epilogue: h2 = tanh(acc+mb2); p[row] = sum_j h2*mw3[j]
    float p[8];
#pragma unroll
    for (int r = 0; r < 8; ++r) {
      float s = 0.f;
#pragma unroll
      for (int nt = 0; nt < 4; ++nt) {
        float h2 = fast_tanh(acc[nt][r] + mb2r[nt]);
        s += h2 * mw3r[nt];
      }
      p[r] = s;
    }
#pragma unroll
    for (int r = 0; r < 8; ++r) {
      p[r] += __shfl_xor(p[r], 1);
      p[r] += __shfl_xor(p[r], 2);
      p[r] += __shfl_xor(p[r], 4);
      p[r] += __shfl_xor(p[r], 8);
    }
    // softplus + partial sum over the 16 s-rows of this tile
    float sh = 0.f;
#pragma unroll
    for (int r = 0; r < 8; ++r) sh += fast_softplus(p[r] + mb3k);
    sh += __shfl_xor(sh, 16);
    if (lane == 0) gsum[((size_t)b * KF + k) * 2 + (t & 1)] = sh;
  }
}

// ---------------------------------------------------------------------------
// Non-mono MLP, fully fused per 16-row b-tile. 8 waves x 4 N-tiles = 512 cols.
// ---------------------------------------------------------------------------
__global__ __launch_bounds__(256) void nonmono_kernel(
    const float* __restrict__ x, const float* __restrict__ nb1,
    const float* __restrict__ nb2, const float* __restrict__ nw3,
    const float* __restrict__ nb3, const _Float16* __restrict__ nw1t,
    const _Float16* __restrict__ nw2t, float* __restrict__ nmout) {
  __shared__ _Float16 a1s[16][NHF];   // 16 KB
  __shared__ float partial[8][16];

  const int b0   = blockIdx.x * 16;
  const int lane = threadIdx.x & 31;
  const int w    = threadIdx.x >> 5;
  const int lh   = lane >> 4;
  const int l15  = lane & 15;
  const int m    = l15;  // A-fragment row

  // ---- stage A: a1 = relu(x_nm @ nw1 + nb1) -> LDS (f16) ----
  v8f acc[4];
#pragma unroll
  for (int nt = 0; nt < 4; ++nt) {
    v8f z = {0.f, 0.f, 0.f, 0.f, 0.f, 0.f, 0.f, 0.f};
    acc[nt] = z;
  }
#pragma unroll
  for (int c = 0; c < 2; ++c) {
    v16h a;
#pragma unroll
    for (int g = 0; g < 2; ++g) {
      int kkb = 32 * c + 16 * lh + 8 * g;
#pragma unroll
      for (int i2 = 0; i2 < 8; ++i2) {
        int kk  = kkb + i2;
        float v = (kk < NMF) ? x[(size_t)(b0 + m) * INF_ + KF + kk] : 0.f;
        a[g * 8 + i2] = (_Float16)v;
      }
    }
#pragma unroll
    for (int nt = 0; nt < 4; ++nt) {
      int j   = 16 * (4 * w + nt) + l15;
      v16h bf = *(const v16h*)(nw1t + (size_t)j * 64 + 32 * c + 16 * lh);
      acc[nt] = wmma_f16(a, bf, acc[nt]);
    }
  }
#pragma unroll
  for (int nt = 0; nt < 4; ++nt) {
    int j      = 16 * (4 * w + nt) + l15;
    float bias = nb1[j];
#pragma unroll
    for (int r = 0; r < 8; ++r) {
      float v = acc[nt][r] + bias;
      v = v > 0.f ? v : 0.f;
      a1s[r + 8 * lh][j] = (_Float16)v;
    }
  }
  __syncthreads();

  // ---- stage B: a2 = relu(a1 @ nw2 + nb2), fused dot with nw3 ----
  v8f acc2[4];
#pragma unroll
  for (int nt = 0; nt < 4; ++nt) {
    v8f z = {0.f, 0.f, 0.f, 0.f, 0.f, 0.f, 0.f, 0.f};
    acc2[nt] = z;
  }
#pragma unroll 4
  for (int c = 0; c < 16; ++c) {
    v8h lo = *(const v8h*)&a1s[m][32 * c + 16 * lh];
    v8h hi = *(const v8h*)&a1s[m][32 * c + 16 * lh + 8];
    v16h a;
#pragma unroll
    for (int i2 = 0; i2 < 8; ++i2) {
      a[i2]     = lo[i2];
      a[8 + i2] = hi[i2];
    }
#pragma unroll
    for (int nt = 0; nt < 4; ++nt) {
      int j   = 16 * (4 * w + nt) + l15;
      v16h bf = *(const v16h*)(nw2t + (size_t)j * NHF + 32 * c + 16 * lh);
      acc2[nt] = wmma_f16(a, bf, acc2[nt]);
    }
  }
  float nb2r[4], nw3r[4];
#pragma unroll
  for (int nt = 0; nt < 4; ++nt) {
    int j    = 16 * (4 * w + nt) + l15;
    nb2r[nt] = nb2[j];
    nw3r[nt] = nw3[j];
  }
  float p[8];
#pragma unroll
  for (int r = 0; r < 8; ++r) {
    float s = 0.f;
#pragma unroll
    for (int nt = 0; nt < 4; ++nt) {
      float v = acc2[nt][r] + nb2r[nt];
      v = v > 0.f ? v : 0.f;
      s += v * nw3r[nt];
    }
    p[r] = s;
  }
#pragma unroll
  for (int r = 0; r < 8; ++r) {
    p[r] += __shfl_xor(p[r], 1);
    p[r] += __shfl_xor(p[r], 2);
    p[r] += __shfl_xor(p[r], 4);
    p[r] += __shfl_xor(p[r], 8);
  }
  if (l15 == 0) {
#pragma unroll
    for (int r = 0; r < 8; ++r) partial[w][8 * lh + r] = p[r];
  }
  __syncthreads();
  if (threadIdx.x < 16) {
    float s = nb3[0];
#pragma unroll
    for (int ww = 0; ww < 8; ++ww) s += partial[ww][threadIdx.x];
    nmout[b0 + threadIdx.x] = s;
  }
}

// ---------------------------------------------------------------------------
// Final combine: out[b] = nm[b] + sum_k mean_s(g) * x_mono * wlin[k]
// ---------------------------------------------------------------------------
__global__ void final_kernel(const float* __restrict__ x,
                             const float* __restrict__ wlin,
                             const float* __restrict__ gsum,
                             const float* __restrict__ nm,
                             float* __restrict__ out) {
  int b = blockIdx.x * blockDim.x + threadIdx.x;
  if (b >= BDIM) return;
  float acc = nm[b];
#pragma unroll
  for (int k = 0; k < KF; ++k) {
    float s = gsum[((size_t)b * KF + k) * 2] + gsum[((size_t)b * KF + k) * 2 + 1];
    acc += (s * (1.0f / (float)SF)) * x[(size_t)b * INF_ + k] * wlin[k];
  }
  out[b] = acc;
}

// ---------------------------------------------------------------------------
extern "C" void kernel_launch(void* const* d_in, const int* in_sizes, int n_in,
                              void* d_out, int out_size, void* d_ws,
                              size_t ws_size, hipStream_t stream) {
  (void)in_sizes; (void)n_in; (void)out_size; (void)ws_size;
  const float* x    = (const float*)d_in[0];
  const float* mw1  = (const float*)d_in[1];
  const float* mb1  = (const float*)d_in[2];
  const float* mw2  = (const float*)d_in[3];
  const float* mb2  = (const float*)d_in[4];
  const float* mw3  = (const float*)d_in[5];
  const float* mb3  = (const float*)d_in[6];
  const float* wlin = (const float*)d_in[7];
  const float* nw1  = (const float*)d_in[8];
  const float* nb1  = (const float*)d_in[9];
  const float* nw2  = (const float*)d_in[10];
  const float* nb2  = (const float*)d_in[11];
  const float* nw3  = (const float*)d_in[12];
  const float* nb3  = (const float*)d_in[13];

  char* ws = (char*)d_ws;
  _Float16* mw2t = (_Float16*)(ws + 0);        // 16*64*64 f16 = 131072 B
  _Float16* nw1t = (_Float16*)(ws + 131072);   // 512*64  f16 =  65536 B
  _Float16* nw2t = (_Float16*)(ws + 196608);   // 512*512 f16 = 524288 B
  float*    gsum = (float*)(ws + 720896);      // 4096*16*2 f32 = 524288 B
  float*    nm   = (float*)(ws + 1245184);     // 4096 f32
  float*    out  = (float*)d_out;

  // 65536 + 32768 + 262144 = 360448 elements -> 1408 blocks of 256
  prep_kernel<<<1408, 256, 0, stream>>>(mw2, nw1, nw2, mw2t, nw1t, nw2t);

  dim3 mg(128, KF);  // 128 blocks/k * 8 waves * 8 tiles = 8192 tiles per k
  mono_kernel<<<mg, 256, 0, stream>>>(x, mw1, mb1, mb2, mw3, mb3, mw2t, gsum);

  nonmono_kernel<<<BDIM / 16, 256, 0, stream>>>(x, nb1, nb2, nw3, nb3, nw1t,
                                                nw2t, nm);

  final_kernel<<<BDIM / 256, 256, 0, stream>>>(x, wlin, gsum, nm, out);
}
